// Qwen3Attention_70248485093534
// MI455X (gfx1250) — matrix operand order repaired and checked by reading, not compiled, not run on hardware
//
#include <hip/hip_runtime.h>

typedef _Float16 half_t;
typedef __attribute__((ext_vector_type(16))) _Float16 v16h;
typedef __attribute__((ext_vector_type(8)))  _Float16 v8h;
typedef __attribute__((ext_vector_type(4)))  _Float16 v4h;
typedef __attribute__((ext_vector_type(8)))  float    v8f;
typedef __attribute__((ext_vector_type(4)))  unsigned int v4u;
typedef __attribute__((ext_vector_type(4)))  int      v4i;
typedef __attribute__((ext_vector_type(8)))  int      v8i;

#define B_    2
#define S_    2048
#define HID_  4096
#define NH_   32
#define NKV_  8
#define D_    128
#define QKV_O 6144   // (NH + 2*NKV) * D

// ---------------------------------------------------------------------------
// fp32 -> f16 elementwise convert (4 elems / thread)
// ---------------------------------------------------------------------------
__global__ __launch_bounds__(256) void k_cvt(const float* __restrict__ s,
                                             half_t* __restrict__ d, long n)
{
  const long i = ((long)blockIdx.x * 256 + threadIdx.x) * 4;
  if (i >= n) return;
  const float4 v = *(const float4*)(s + i);
  v4h o;
  o[0] = (half_t)v.x; o[1] = (half_t)v.y; o[2] = (half_t)v.z; o[3] = (half_t)v.w;
  *(v4h*)(d + i) = o;
}

// ---------------------------------------------------------------------------
// TDM: DMA a [rows x rowlen] f16 tile (row stride = stride_elems) into LDS.
// Descriptor built from wave-uniform scalars -> lives in SGPRs.
// ---------------------------------------------------------------------------
__device__ __forceinline__ void tdm_load_tile(unsigned lds_off,
                                              const half_t* gaddr,
                                              unsigned rows, unsigned rowlen,
                                              unsigned tensor_d0, unsigned tensor_d1,
                                              unsigned stride_elems)
{
  const unsigned long long ga = (unsigned long long)(uintptr_t)gaddr;
  v4u g0;
  g0[0] = 1u;                                           // count=1 (valid user D#)
  g0[1] = lds_off;                                      // lds_addr (bytes)
  g0[2] = (unsigned)(ga & 0xffffffffu);                 // global_addr[31:0]
  g0[3] = (unsigned)((ga >> 32) & 0x01ffffffu)          // global_addr[56:32]
        | (2u << 30);                                   // type = 2 ("image")
  v8i g1;
  g1[0] = (int)(1u << 16);                              // data_size=1 -> 2B elems
  g1[1] = (int)((tensor_d0 & 0xffffu) << 16);           // tensor_dim0[15:0]
  g1[2] = (int)(((tensor_d0 >> 16) & 0xffffu)
        | ((tensor_d1 & 0xffffu) << 16));               // dim0[31:16] | dim1[15:0]
  g1[3] = (int)(((tensor_d1 >> 16) & 0xffffu)
        | ((rowlen & 0xffffu) << 16));                  // dim1[31:16] | tile_dim0
  g1[4] = (int)(rows & 0xffffu);                        // tile_dim1 (tile_dim2=0)
  g1[5] = (int)stride_elems;                            // tensor_dim0_stride[31:0]
  g1[6] = 0;                                            // stride hi | dim1_stride lo
  g1[7] = 0;
  const v4i gz = {0, 0, 0, 0};
#if defined(__clang_major__) && (__clang_major__ >= 23)
  const v8i gz8 = {0, 0, 0, 0, 0, 0, 0, 0};
  __builtin_amdgcn_tensor_load_to_lds(g0, g1, gz, gz, gz8, 0);
#else
  __builtin_amdgcn_tensor_load_to_lds(g0, g1, gz, gz, 0);
#endif
}

// ---------------------------------------------------------------------------
// C[M][N] = A[M][K] * B[N][K]^T   (A,B f16, C f32)
// Workgroup tile 64M x 128N, 8 waves. B tile (128 x 32) staged into LDS by the
// Tensor Data Mover, double-buffered: TDM for step k+1 overlaps WMMAs of step k.
// A-frags load straight from global (rows are per-lane unique; L2-resident).
// ---------------------------------------------------------------------------
__global__ __launch_bounds__(256) void k_gemm_nt(const half_t* __restrict__ A,
                                                 const half_t* __restrict__ B,
                                                 float* __restrict__ C,
                                                 int M, int N, int K)
{
  __shared__ __align__(32) half_t bstage[2][128 * 32];   // 2 x 8KB
  const int wave = threadIdx.x >> 5;
  const int lane = threadIdx.x & 31;
  const int lo = lane & 15, hi = lane >> 4;
  const long Mb = (long)blockIdx.y * 64 + (wave & 3) * 16;
  const long Nb = (long)blockIdx.x * 128;
  const int  nw = (wave >> 2) * 64;        // this wave's N offset inside the stage
  const half_t* Arow = A + (Mb + lo) * (long)K;
  const half_t* Btile = B + Nb * (long)K;

  const v8f vz = {0.f,0.f,0.f,0.f,0.f,0.f,0.f,0.f};
  v8f acc[4];
#pragma unroll
  for (int t = 0; t < 4; ++t) acc[t] = vz;

  const bool issuer = (wave == 0);
  if (issuer)
    tdm_load_tile((unsigned)(uintptr_t)&bstage[0][0], Btile, 128, 32,
                  (unsigned)K, (unsigned)N, (unsigned)K);

  int buf = 0;
  for (int k0 = 0; k0 < K; k0 += 32, buf ^= 1) {
    if (issuer) __builtin_amdgcn_s_wait_tensorcnt((short)0);
    __syncthreads();                                    // stage[buf] ready for all
    if (issuer && (k0 + 32) < K)                        // prefetch next K-step
      tdm_load_tile((unsigned)(uintptr_t)&bstage[buf ^ 1][0], Btile + (k0 + 32),
                    128, 32, (unsigned)K, (unsigned)N, (unsigned)K);

    // 16-bit A 16x32 frag: 16 contiguous K @ hi*16
    union { v16h v; v8h h8[2]; } ua;
    ua.h8[0] = *(const v8h*)(Arow + k0 + hi * 16);
    ua.h8[1] = *(const v8h*)(Arow + k0 + hi * 16 + 8);
    const v16h af = ua.v;
    const half_t* bs = &bstage[buf][0];
#pragma unroll
    for (int t = 0; t < 4; ++t) {
      // B 32x16 frag from LDS: lane -> N row (nw+t*16+lo), 16 contiguous K @ hi*16
      const v16h bf = *(const v16h*)(bs + (nw + t * 16 + lo) * 32 + hi * 16);
      acc[t] = __builtin_amdgcn_wmma_f32_16x16x32_f16(false, af, false, bf,
                                                      (short)0, acc[t], false, false);
    }
    __syncthreads();                     // all reads of stage[buf] done before reuse
  }

  // f32 C 16x16 layout: row = r + 8*hi, col = lo
#pragma unroll
  for (int t = 0; t < 4; ++t)
#pragma unroll
    for (int r = 0; r < 8; ++r)
      C[(Mb + r + 8 * hi) * (long)N + Nb + nw + t * 16 + lo] = acc[t][r];
}

// ---------------------------------------------------------------------------
// RMSNorm + RoPE (+ V transpose). One wave per (token, head-slot).
// slots: 0..31 -> Q heads, 32..39 -> K heads, 40..47 -> V heads.
// Q -> [b][h][s][d] f16 ; K -> [b][hk][s][d] f16 ; V -> transposed [b][hk][d][s] f16
// ---------------------------------------------------------------------------
__global__ __launch_bounds__(256) void k_normrope(const float* __restrict__ qkv,
                                                  const int* __restrict__ pos,
                                                  const float* __restrict__ qw,
                                                  const float* __restrict__ kw,
                                                  half_t* __restrict__ Qo,
                                                  half_t* __restrict__ Ko,
                                                  half_t* __restrict__ Vto)
{
  const int  lane  = threadIdx.x & 31;
  const long gw    = (long)blockIdx.x * 8 + (threadIdx.x >> 5);
  const long token = gw / 48;
  const int  slot  = (int)(gw % 48);
  const int  b     = (int)(token / S_);
  const int  s     = (int)(token % S_);

  const float4 xv = *(const float4*)(qkv + token * QKV_O + (long)slot * D_ + lane * 4);
  float x[4] = {xv.x, xv.y, xv.z, xv.w};

  if (slot >= 40) {                      // V: plain convert, transposed store
    const int hv = slot - 40;
    half_t* vp = Vto + ((long)(b * NKV_ + hv) * D_ + lane * 4) * S_ + s;
#pragma unroll
    for (int i = 0; i < 4; ++i) vp[(long)i * S_] = (half_t)x[i];
    return;
  }

  // RMS norm over D=128 (32 lanes * 4)
  float ss = x[0]*x[0] + x[1]*x[1] + x[2]*x[2] + x[3]*x[3];
#pragma unroll
  for (int m = 16; m >= 1; m >>= 1) ss += __shfl_xor(ss, m, 32);
  const float inv = rsqrtf(ss * (1.0f / 128.0f) + 1e-6f);

  const bool isq = (slot < 32);
  const float* w = isq ? qw : kw;
  const int p = pos[token];

  float xn[4];
#pragma unroll
  for (int i = 0; i < 4; ++i) xn[i] = x[i] * inv * w[lane * 4 + i];

  // RoPE: pair (d, d+64); partner lives in lane^16
  float outv[4];
#pragma unroll
  for (int i = 0; i < 4; ++i) {
    const int d = lane * 4 + i;
    const int j = d & 63;
    const float invf = __powf(10000.0f, -(float)j * (1.0f / 64.0f));
    const float ang  = (float)p * invf;
    const float c = __cosf(ang), sn = __sinf(ang);
    const float partner = __shfl_xor(xn[i], 16, 32);
    outv[i] = (lane < 16) ? (xn[i] * c - partner * sn)
                          : (xn[i] * c + partner * sn);
  }

  if (isq) {
    half_t* qp = Qo + ((long)(b * NH_ + slot) * S_ + s) * D_ + lane * 4;
#pragma unroll
    for (int i = 0; i < 4; ++i) qp[i] = (half_t)outv[i];
  } else {
    const int hk = slot - 32;
    half_t* kp = Ko + ((long)(b * NKV_ + hk) * S_ + s) * D_ + lane * 4;
#pragma unroll
    for (int i = 0; i < 4; ++i) kp[i] = (half_t)outv[i];
  }
}

// ---------------------------------------------------------------------------
// Flash attention, causal, GQA (rep=4). Block = 8 waves; wave owns 16 queries.
// grid = (B*NH, S/128). Scores: QK^T via WMMA; online softmax with 16-lane
// shuffle reductions; P goes through per-wave LDS to get A-frag layout; PV
// via WMMA into 16x128 f32 accumulator. Output: attn [b][s][h*D+d] f16.
// ---------------------------------------------------------------------------
__global__ __launch_bounds__(256) void k_attn(const half_t* __restrict__ Q,
                                              const half_t* __restrict__ Kc,
                                              const half_t* __restrict__ Vt,
                                              half_t* __restrict__ O)
{
  __shared__ __align__(32) half_t p_lds[8][16 * 32];   // per-wave 16x32 P tile
  const int wave = threadIdx.x >> 5;
  const int lane = threadIdx.x & 31;
  const int lo = lane & 15, hi = lane >> 4;
  const int bh = blockIdx.x;
  const int b = bh / NH_, h = bh % NH_;
  const int hk = h / (NH_ / NKV_);
  const int qbase = blockIdx.y * 128 + wave * 16;

  const half_t* Qp = Q  + ((long)(b * NH_  + h ) * S_ + qbase) * D_;
  const half_t* Kp = Kc +  (long)(b * NKV_ + hk) * S_ * D_;
  const half_t* Vp = Vt +  (long)(b * NKV_ + hk) * D_ * S_;

  // Q A-frags for all 4 d-chunks, resident for the whole key loop
  v16h qa[4];
  const half_t* qrow = Qp + (long)lo * D_;
#pragma unroll
  for (int t = 0; t < 4; ++t) {
    union { v16h v; v8h h8[2]; } u;
    u.h8[0] = *(const v8h*)(qrow + t * 32 + hi * 16);
    u.h8[1] = *(const v8h*)(qrow + t * 32 + hi * 16 + 8);
    qa[t] = u.v;
  }

  const v8f vz = {0.f,0.f,0.f,0.f,0.f,0.f,0.f,0.f};
  v8f acc[8];
#pragma unroll
  for (int t = 0; t < 8; ++t) acc[t] = vz;
  float mrow[8], lrow[8];
#pragma unroll
  for (int r = 0; r < 8; ++r) { mrow[r] = -1.0e30f; lrow[r] = 0.0f; }

  const float scale = 0.08838834764831845f;   // 1/sqrt(128)
  half_t* pl = &p_lds[wave][0];
  const int qmax = qbase + 15;

  for (int kb = 0; kb <= qmax; kb += 32) {
    // ---- scores: two 16x16 tiles (keys kb..+15, kb+16..+31) over d=128
    v8f s0 = vz, s1 = vz;
#pragma unroll
    for (int t = 0; t < 4; ++t) {
      const v16h b0 = *(const v16h*)(Kp + (long)(kb + lo)      * D_ + t * 32 + hi * 16);
      const v16h b1 = *(const v16h*)(Kp + (long)(kb + 16 + lo) * D_ + t * 32 + hi * 16);
      s0 = __builtin_amdgcn_wmma_f32_16x16x32_f16(false, qa[t], false, b0, (short)0, s0, false, false);
      s1 = __builtin_amdgcn_wmma_f32_16x16x32_f16(false, qa[t], false, b1, (short)0, s1, false, false);
    }
    // ---- causal mask + online softmax (row M = r + 8*hi, col N = lo)
#pragma unroll
    for (int r = 0; r < 8; ++r) {
      const int qg = qbase + r + 8 * hi;
      float v0 = s0[r] * scale; if (kb + lo      > qg) v0 = -1.0e30f;
      float v1 = s1[r] * scale; if (kb + 16 + lo > qg) v1 = -1.0e30f;
      float rmax = fmaxf(v0, v1);
      rmax = fmaxf(rmax, __shfl_xor(rmax, 1, 32));
      rmax = fmaxf(rmax, __shfl_xor(rmax, 2, 32));
      rmax = fmaxf(rmax, __shfl_xor(rmax, 4, 32));
      rmax = fmaxf(rmax, __shfl_xor(rmax, 8, 32));
      const float mnew = fmaxf(mrow[r], rmax);
      const float corr = __expf(mrow[r] - mnew);
      const float p0 = __expf(v0 - mnew);
      const float p1 = __expf(v1 - mnew);
      float rsum = p0 + p1;
      rsum += __shfl_xor(rsum, 1, 32);
      rsum += __shfl_xor(rsum, 2, 32);
      rsum += __shfl_xor(rsum, 4, 32);
      rsum += __shfl_xor(rsum, 8, 32);
      mrow[r] = mnew;
      lrow[r] = lrow[r] * corr + rsum;
#pragma unroll
      for (int t = 0; t < 8; ++t) acc[t][r] *= corr;
      pl[(r + 8 * hi) * 32 + lo]      = (half_t)p0;
      pl[(r + 8 * hi) * 32 + 16 + lo] = (half_t)p1;
    }
    // per-wave LDS RAW fence (no workgroup barrier: waves have unequal trip counts)
    asm volatile("s_wait_dscnt 0" ::: "memory");
    // re-read P in A-frag (16x32) layout
    union { v16h v; v8h h8[2]; } up;
    up.h8[0] = *(const v8h*)(pl + lo * 32 + hi * 16);
    up.h8[1] = *(const v8h*)(pl + lo * 32 + hi * 16 + 8);
    const v16h pa = up.v;
    // ---- P·V : B-frag from transposed V, contiguous over keys
#pragma unroll
    for (int t = 0; t < 8; ++t) {
      const v16h vb = *(const v16h*)(Vp + (long)(t * 16 + lo) * S_ + kb + hi * 16);
      acc[t] = __builtin_amdgcn_wmma_f32_16x16x32_f16(false, pa, false, vb, (short)0, acc[t], false, false);
    }
  }

  // normalize and store attn output as f16, layout [b][s][h*D + d]
#pragma unroll
  for (int t = 0; t < 8; ++t)
#pragma unroll
    for (int r = 0; r < 8; ++r) {
      const int qg = qbase + r + 8 * hi;
      const float o = acc[t][r] / lrow[r];
      O[((long)b * S_ + qg) * (NH_ * D_) + h * D_ + t * 16 + lo] = (half_t)o;
    }
}

// ---------------------------------------------------------------------------
extern "C" void kernel_launch(void* const* d_in, const int* in_sizes, int n_in,
                              void* d_out, int out_size, void* d_ws, size_t ws_size,
                              hipStream_t stream)
{
  (void)in_sizes; (void)n_in; (void)out_size; (void)ws_size;
  const int*   pos  = (const int*)d_in[0];
  const float* hid  = (const float*)d_in[1];
  const float* wqkv = (const float*)d_in[2];
  const float* wo   = (const float*)d_in[3];
  const float* qnw  = (const float*)d_in[4];
  const float* knw  = (const float*)d_in[5];
  float* out = (float*)d_out;

  char* ws = (char*)d_ws;
  size_t off = 0;
  auto take = [&](size_t bytes) -> char* {
    char* p = ws + off;
    off = (off + bytes + 255) & ~(size_t)255;
    return p;
  };
  half_t* hid_h  = (half_t*)take((size_t)B_ * S_ * HID_ * 2);
  half_t* wqkv_h = (half_t*)take((size_t)QKV_O * HID_ * 2);
  half_t* wo_h   = (half_t*)take((size_t)HID_ * NH_ * D_ * 2);
  float*  qkv_f  = (float*) take((size_t)B_ * S_ * QKV_O * 4);
  half_t* Qh     = (half_t*)take((size_t)B_ * NH_  * S_ * D_ * 2);
  half_t* Kh     = (half_t*)take((size_t)B_ * NKV_ * S_ * D_ * 2);
  half_t* Vth    = (half_t*)take((size_t)B_ * NKV_ * S_ * D_ * 2);
  half_t* attn_h = (half_t*)take((size_t)B_ * S_ * NH_ * D_ * 2);

  const long n1 = (long)B_ * S_ * HID_;     // hidden elems
  const long n2 = (long)QKV_O * HID_;       // w_qkv elems
  const long n3 = (long)HID_ * NH_ * D_;    // w_o elems
  k_cvt<<<(int)(n1 / 1024), 256, 0, stream>>>(hid,  hid_h,  n1);
  k_cvt<<<(int)(n2 / 1024), 256, 0, stream>>>(wqkv, wqkv_h, n2);
  k_cvt<<<(int)(n3 / 1024), 256, 0, stream>>>(wo,   wo_h,   n3);

  // QKV projection: [4096 x 6144] = hid[4096 x 4096] * w_qkv[6144 x 4096]^T
  dim3 g1(QKV_O / 128, (B_ * S_) / 64);
  k_gemm_nt<<<g1, 256, 0, stream>>>(hid_h, wqkv_h, qkv_f, B_ * S_, QKV_O, HID_);

  // norm + rope + V transpose: one wave per (token, slot), 8 waves/block
  const long waves = (long)B_ * S_ * 48;
  k_normrope<<<(int)(waves / 8), 256, 0, stream>>>(qkv_f, pos, qnw, knw, Qh, Kh, Vth);

  // flash attention
  dim3 ga(B_ * NH_, S_ / 128);
  k_attn<<<ga, 256, 0, stream>>>(Qh, Kh, Vth, attn_h);

  // output projection: [4096 x 4096] = attn[4096 x 4096] * w_o[4096 x 4096]^T
  dim3 g2(HID_ / 128, (B_ * S_) / 64);
  k_gemm_nt<<<g2, 256, 0, stream>>>(attn_h, wo_h, out, B_ * S_, HID_, NH_ * D_);
}
